// GRU_7215545057457
// MI455X (gfx1250) — compile-verified, matrix operand order checked by reading, not run on hardware
//
#include <hip/hip_runtime.h>
#include <hip/hip_bf16.h>

// GRU scan, N=524288 rows, H=64, T=24, via bf16 WMMA (v_wmma_f32_16x16x32_bf16).
// One wave32 owns 16 rows for the whole time loop; h stays register-resident in
// WMMA C-layout. LDS is used (wave-private, barrier-free) for the C->A layout
// transpose between the two GEMMs / between timesteps, and (block-shared) for
// the bf16-converted weight matrices.

#define HGRU 64
#define TSTEPS 24
#define ROWS_PER_WAVE 16
#define WAVES_PER_BLK 8
#define BLK_THREADS 256
#define WSTRIDE 72   // bf16 elements per LDS row (144B) -> bank-conflict-free b128 reads

typedef __attribute__((ext_vector_type(16))) __bf16 v16bf;
typedef __attribute__((ext_vector_type(8)))  float  v8f;

__device__ __forceinline__ v16bf ld_mat16(const __bf16* p0, const __bf16* p1) {
  v16bf r;
  *reinterpret_cast<uint4*>(&r)       = *reinterpret_cast<const uint4*>(p0);
  *(reinterpret_cast<uint4*>(&r) + 1) = *reinterpret_cast<const uint4*>(p1);
  return r;
}

__device__ __forceinline__ float fast_sigmoid(float z) {
  return __builtin_amdgcn_rcpf(1.0f + __expf(-z));
}
__device__ __forceinline__ float fast_tanh(float z) {
  float e = __expf(-2.0f * z);
  return (1.0f - e) * __builtin_amdgcn_rcpf(1.0f + e);
}

__global__ __launch_bounds__(BLK_THREADS) void gru_scan_wmma(
    const float* __restrict__ x,  const float* __restrict__ W1,
    const float* __restrict__ b1, const float* __restrict__ W2,
    const float* __restrict__ b2, float* __restrict__ out) {
  // rows 0..127: W1[:,1:65] as bf16 ; rows 128..191: W2[:,1:65] as bf16
  __shared__ alignas(16) __bf16 wlds[192][WSTRIDE];
  // per-wave staging tile for h / (r*h) in A-operand-readable row-major form
  __shared__ alignas(16) __bf16 hlds[WAVES_PER_BLK][16][WSTRIDE];
  __shared__ float xlds[WAVES_PER_BLK][16];

  const int tid = threadIdx.x;
  for (int idx = tid; idx < 192 * 64; idx += BLK_THREADS) {
    int row = idx >> 6, k = idx & 63;
    float v = (row < 128) ? W1[row * 65 + 1 + k] : W2[(row - 128) * 65 + 1 + k];
    wlds[row][k] = (__bf16)v;
  }
  __syncthreads();

  const int w    = tid >> 5;
  const int lane = tid & 31;
  const int l16  = lane & 15;
  const int hi   = lane >> 4;   // which half-wave (M/K split in WMMA layouts)
  const long rowbase = (long)(blockIdx.x * WAVES_PER_BLK + w) * ROWS_PER_WAVE;

  // Rank-1 x-term coefficients and biases: step-invariant, lane-resident.
  float b1r[8], w10[8], b2r[4], w20[4];
#pragma unroll
  for (int j = 0; j < 8; ++j) { b1r[j] = b1[j * 16 + l16]; w10[j] = W1[(j * 16 + l16) * 65]; }
#pragma unroll
  for (int j = 0; j < 4; ++j) { b2r[j] = b2[j * 16 + l16]; w20[j] = W2[(j * 16 + l16) * 65]; }

  // h in WMMA C-layout: h[jc][v] = row (v + 8*hi), col (jc*16 + l16), f32.
  float h[4][8];
#pragma unroll
  for (int jc = 0; jc < 4; ++jc)
#pragma unroll
    for (int v = 0; v < 8; ++v) h[jc][v] = 0.0f;

  for (int t = 0; t < TSTEPS; ++t) {
    // Stage x_t for the 16 rows (wave-private; intra-wave LDS is in-order).
    if (hi == 0) xlds[w][l16] = x[(rowbase + l16) * TSTEPS + t];
    float xv[8];
#pragma unroll
    for (int v = 0; v < 8; ++v) xv[v] = xlds[w][v + hi * 8];

    // Stage h as bf16 row-major for A-operand loads.
#pragma unroll
    for (int jc = 0; jc < 4; ++jc)
#pragma unroll
      for (int v = 0; v < 8; ++v)
        hlds[w][v + hi * 8][jc * 16 + l16] = (__bf16)h[jc][v];

    const __bf16* hrow = &hlds[w][l16][0];
    // A operand (16x32 bf16) per ISA layout: lanes<16 -> K{0..7,16..23}, lanes>=16 -> K{8..15,24..31}
    v16bf A0 = ld_mat16(hrow + hi * 8,      hrow + 16 + hi * 8);
    v16bf A1 = ld_mat16(hrow + 32 + hi * 8, hrow + 48 + hi * 8);

    // GEMM1: ru = sigmoid(h @ W1h^T + x*W1[:,0] + b1), 8 col-tiles of 16.
    float u[4][8];
#pragma unroll
    for (int j = 0; j < 8; ++j) {
      const __bf16* brow = &wlds[j * 16 + l16][0];
      v16bf B0 = ld_mat16(brow + hi * 8,      brow + 16 + hi * 8);
      v16bf B1 = ld_mat16(brow + 32 + hi * 8, brow + 48 + hi * 8);
      v8f acc;
#pragma unroll
      for (int v = 0; v < 8; ++v) acc[v] = b1r[j] + xv[v] * w10[j];
      acc = __builtin_amdgcn_wmma_f32_16x16x32_bf16(false, A0, false, B0, (short)0, acc, false, false);
      acc = __builtin_amdgcn_wmma_f32_16x16x32_bf16(false, A1, false, B1, (short)0, acc, false, false);
      if (j < 4) {
        // r tile -> stage (r*h) bf16 into the same LDS tile (A regs already loaded)
#pragma unroll
        for (int v = 0; v < 8; ++v) {
          float r = fast_sigmoid(acc[v]);
          hlds[w][v + hi * 8][j * 16 + l16] = (__bf16)(r * h[j][v]);
        }
      } else {
#pragma unroll
        for (int v = 0; v < 8; ++v) u[j - 4][v] = fast_sigmoid(acc[v]);
      }
    }

    // A operand for GEMM2 = (r*h)
    v16bf A0r = ld_mat16(hrow + hi * 8,      hrow + 16 + hi * 8);
    v16bf A1r = ld_mat16(hrow + 32 + hi * 8, hrow + 48 + hi * 8);

    // GEMM2: c = tanh((r*h) @ W2h^T + x*W2[:,0] + b2); h = u*h + (1-u)*c
#pragma unroll
    for (int j = 0; j < 4; ++j) {
      const __bf16* brow = &wlds[128 + j * 16 + l16][0];
      v16bf B0 = ld_mat16(brow + hi * 8,      brow + 16 + hi * 8);
      v16bf B1 = ld_mat16(brow + 32 + hi * 8, brow + 48 + hi * 8);
      v8f acc;
#pragma unroll
      for (int v = 0; v < 8; ++v) acc[v] = b2r[j] + xv[v] * w20[j];
      acc = __builtin_amdgcn_wmma_f32_16x16x32_bf16(false, A0r, false, B0, (short)0, acc, false, false);
      acc = __builtin_amdgcn_wmma_f32_16x16x32_bf16(false, A1r, false, B1, (short)0, acc, false, false);
#pragma unroll
      for (int v = 0; v < 8; ++v) {
        float c = fast_tanh(acc[v]);
        h[j][v] = u[j][v] * h[j][v] + (1.0f - u[j][v]) * c;
      }
    }
  }

  // Write final h (f32, row-major N x 64). Each (v,hi) pair: 16 lanes x 4B contiguous.
#pragma unroll
  for (int jc = 0; jc < 4; ++jc)
#pragma unroll
    for (int v = 0; v < 8; ++v)
      out[(rowbase + v + hi * 8) * HGRU + jc * 16 + l16] = h[jc][v];
}

extern "C" void kernel_launch(void* const* d_in, const int* in_sizes, int n_in,
                              void* d_out, int out_size, void* d_ws, size_t ws_size,
                              hipStream_t stream) {
  const float* x  = (const float*)d_in[0];
  // d_in[1] (edge_index) and d_in[2] (edge_weight) are unused by the reference.
  const float* W1 = (const float*)d_in[3];
  const float* b1 = (const float*)d_in[4];
  const float* W2 = (const float*)d_in[5];
  const float* b2 = (const float*)d_in[6];
  float* out = (float*)d_out;

  const int n_rows = in_sizes[0] / TSTEPS;                    // 524288
  const int grid   = n_rows / (ROWS_PER_WAVE * WAVES_PER_BLK); // 4096
  gru_scan_wmma<<<grid, BLK_THREADS, 0, stream>>>(x, W1, b1, W2, b2, out);
}
